// ImageChebychevConvolution_79602923864566
// MI455X (gfx1250) — matrix operand order repaired and checked by reading, not compiled, not run on hardware
//
#include <hip/hip_runtime.h>

// ImageChebychevConvolution on MI455X (gfx1250), wave32, WMMA bf16 path.
//
// out[b,u,p,q] = sum_{k1,k2,c,h,w} coefs[k1,k2,c,u] cheb1[k1,h,p] x[b,c,h,w] cheb2[k2,w,q]
// B=32, C=U=16, LEN=128, POWERS=5.
//
// Pipeline (all matmuls via v_wmma_f32_16x16x32_bf16, f32 accumulate):
//   prep   : bf16 tables  cheb1T[k][p][h], cheb2b[k][w][q], coefsA[(k2,u)][(k1,c)]
//   stage1 : t1[b,(k1,c),p,w] = sum_h cheb1T[k1][p][h] * x[b,c,h,w]      (bf16 ws)
//            (cheb1T chunk staged with global_load_async_to_lds_b128)
//   stage2 : s[b,(k2,u),p,w]  = sum_{k1c} coefsA[(k2u)][(k1c)] * t1      (in-place ws)
//   stage3 : out[b,u,p,q]     = sum_{k2,w} s[b,(k2,u),p,w] * cheb2b[k2][w][q]
//            (s chunk staged as one 3D-tile tensor_load_to_lds per K step)
//
// The 84 MB bf16 intermediate stays resident in the 192 MB L2; HBM traffic is
// essentially x-in + out-out (67 MB) -> memory-bound, bf16 is the right precision.

#define B_   32
#define C_   16
#define U_   16
#define LEN_ 128
#define POW_ 5
#define KC_  80     // k1*16+c  == k2*16+u  contraction width
#define IMG_ (LEN_ * LEN_)

typedef __attribute__((ext_vector_type(16))) __bf16 v16bf;
typedef __attribute__((ext_vector_type(8)))  float  v8f;
typedef __attribute__((ext_vector_type(4)))  unsigned v4u;
typedef __attribute__((ext_vector_type(8)))  unsigned v8u;

static __device__ inline __bf16 bfc(float f) { return (__bf16)f; }

static __device__ inline v8f v8f_zero() {
  v8f z = {0.f, 0.f, 0.f, 0.f, 0.f, 0.f, 0.f, 0.f};
  return z;
}

// Low 32 bits of a flat LDS pointer == workgroup-relative LDS byte offset
// (LDS aperture maps addr[31:0] directly; ISA 10.2).
static __device__ inline unsigned lds_off(const void* p) {
  return (unsigned)(size_t)p;
}

// --- WMMA fragment loaders (k = 16*(lane/16) + element, same order for A and B, wave32) -----------------
// A-matrix 16x32 bf16, source row-major [m][ld] (k contiguous).
static __device__ inline v16bf load_a(const __bf16* a, int ld, int lane) {
  v16bf f;
  int m  = lane & 15;
  int kh = (lane >> 4) & 1;
  const __bf16* row = a + m * ld;
#pragma unroll
  for (int v = 0; v < 8; ++v) {
    int k0 = kh * 16 + 2 * v;
    f[2 * v]     = row[k0];
    f[2 * v + 1] = row[k0 + 1];
  }
  return f;
}

// B-matrix 32x16 bf16, source is B^T row-major [n][ld] (k contiguous).
static __device__ inline v16bf load_bt(const __bf16* bt, int ld, int lane) {
  v16bf f;
  int n    = lane & 15;
  int half = (lane >> 4) & 1;
  const __bf16* row = bt + n * ld;
#pragma unroll
  for (int v = 0; v < 8; ++v) {
    int k0 = half * 16 + 2 * v;
    f[2 * v]     = row[k0];
    f[2 * v + 1] = row[k0 + 1];
  }
  return f;
}

static __device__ inline v8f wmma_bf16(v16bf a, v16bf b, v8f c) {
  return __builtin_amdgcn_wmma_f32_16x16x32_bf16(false, a, false, b,
                                                 (short)0, c, false, false);
}

// --- prep: build bf16 tables -------------------------------------------------
__global__ void __launch_bounds__(256)
prep_kernel(const float* __restrict__ cheb1, const float* __restrict__ cheb2,
            const float* __restrict__ coefs,
            __bf16* __restrict__ cheb1T, __bf16* __restrict__ cheb2b,
            __bf16* __restrict__ coefsA) {
  int tid    = blockIdx.x * blockDim.x + threadIdx.x;
  int stride = gridDim.x * blockDim.x;
  const int NCH = POW_ * IMG_;
  for (int i = tid; i < NCH; i += stride) {
    int k = i / IMG_, rem = i % IMG_;
    int p = rem / LEN_, h = rem % LEN_;
    cheb1T[i] = bfc(cheb1[k * IMG_ + h * LEN_ + p]);  // transpose h<->p
    cheb2b[i] = bfc(cheb2[i]);                        // [k][w][q] as-is
  }
  for (int i = tid; i < KC_ * KC_; i += stride) {
    int m = i / KC_, k = i % KC_;        // m = k2*16+u, k = k1*16+c
    int k2 = m / 16, u = m % 16, k1 = k / 16, c = k % 16;
    coefsA[i] = bfc(coefs[((k1 * POW_ + k2) * C_ + c) * U_ + u]);
  }
}

// --- stage 1: t1[b,(k1,c)] = cheb1T[k1] x X[b,c]  (128x128x128 GEMM) ---------
__global__ void __launch_bounds__(256)
stage1_kernel(const float* __restrict__ x, const __bf16* __restrict__ cheb1T,
              __bf16* __restrict__ t1) {
  __shared__ __bf16 sA[128 * 32];   // cheb1T chunk: [p][h_local] (async copy)
  __shared__ __bf16 sBT[128 * 32];  // X^T chunk:    [w][h_local] (f32->bf16)
  int bid = blockIdx.x;
  int c   = bid % C_;
  int k1  = (bid / C_) % POW_;
  int b   = bid / (C_ * POW_);
  int tid = threadIdx.x, lane = tid & 31, wv = tid >> 5;  // 8 waves

  const float*  xg = x + (size_t)(b * C_ + c) * IMG_;
  const __bf16* ag = cheb1T + (size_t)k1 * IMG_;

  v8f acc[8];
#pragma unroll
  for (int q = 0; q < 8; ++q) acc[q] = v8f_zero();

  for (int h0 = 0; h0 < LEN_; h0 += 32) {
    __syncthreads();
    // sA: 128 rows x 64B, bf16->bf16: async copy, 512 x 16B chunks.
    {
      int ci = tid;
#pragma unroll
      for (int it = 0; it < 2; ++it, ci += 256) {
        int p = ci >> 2, cb = (ci & 3) * 16;
        const char* gp = (const char*)(ag + p * LEN_ + h0) + cb;
        unsigned la = lds_off(&sA[0]) + ci * 16;
        asm volatile("global_load_async_to_lds_b128 %0, %1, off"
                     :: "v"(la), "v"(gp) : "memory");
      }
    }
    // sBT: f32->bf16 transposing copy (must stay in VALU).
    for (int i = tid; i < 128 * 32; i += 256) {
      int hh = i >> 7, w = i & 127;
      sBT[w * 32 + hh] = bfc(xg[(h0 + hh) * LEN_ + w]);
    }
    asm volatile("s_wait_asynccnt 0x0" ::: "memory");
    __syncthreads();
    v16bf af = load_a(&sA[(wv * 16) * 32], 32, lane);  // wave's p-tile rows
#pragma unroll
    for (int qt = 0; qt < 8; ++qt) {
      v16bf bfr = load_bt(&sBT[(qt * 16) * 32], 32, lane);
      acc[qt] = wmma_bf16(af, bfr, acc[qt]);
    }
  }
  // D layout: n = lane&15, m = r + 8*(lane>>4)
  __bf16* og = t1 + (size_t)(b * KC_ + k1 * 16 + c) * IMG_;
  int n = lane & 15, half = lane >> 4;
#pragma unroll
  for (int qt = 0; qt < 8; ++qt)
#pragma unroll
    for (int r = 0; r < 8; ++r)
      og[(wv * 16 + r + 8 * half) * LEN_ + qt * 16 + n] = bfc(acc[qt][r]);
}

// --- stage 2 (in-place): s[(k2u), j] = coefsA[(k2u)][(k1c)] x t1[(k1c), j] ---
__global__ void __launch_bounds__(256)
stage2_kernel(__bf16* __restrict__ t1, const __bf16* __restrict__ coefsA) {
  __shared__ __bf16 sT[256 * 96];  // t^T: [j=pixel][k1c padded 80->96]
  __shared__ __bf16 sC[80 * 96];   // A:   [k2u][k1c padded 80->96]
  int bid = blockIdx.x;
  int wt = bid & 7, pt = (bid >> 3) & 7, b = bid >> 6;
  int tid = threadIdx.x, lane = tid & 31, wv = tid >> 5;

  for (int i = tid; i < 80 * 96; i += 256) {
    int m = i / 96, k = i % 96;
    sC[i] = (k < 80) ? coefsA[m * KC_ + k] : bfc(0.0f);
  }
  __bf16* tg = t1 + (size_t)b * KC_ * IMG_;
  for (int i = tid; i < 80 * 256; i += 256) {
    int kc = i >> 8, j = i & 255;
    int pl = j >> 4, wl = j & 15;
    sT[j * 96 + kc] =
        tg[(size_t)kc * IMG_ + (pt * 16 + pl) * LEN_ + wt * 16 + wl];
  }
  for (int i = tid; i < 256 * 16; i += 256) {
    int j = i >> 4, k = 80 + (i & 15);
    sT[j * 96 + k] = bfc(0.0f);
  }
  __syncthreads();

  int n = lane & 15, half = lane >> 4;
#pragma unroll
  for (int t = 0; t < 10; ++t) {  // 80 output tiles / 8 waves
    int tileid = wv * 10 + t;
    int mt = tileid >> 4, nt = tileid & 15;
    v8f acc = v8f_zero();
#pragma unroll
    for (int kc = 0; kc < 3; ++kc) {  // K = 96 (padded)
      v16bf af  = load_a(&sC[(mt * 16) * 96 + kc * 32], 96, lane);
      v16bf bfr = load_bt(&sT[(nt * 16) * 96 + kc * 32], 96, lane);
      acc = wmma_bf16(af, bfr, acc);
    }
    int j  = nt * 16 + n;
    int pl = j >> 4, wl = j & 15;
#pragma unroll
    for (int r = 0; r < 8; ++r) {
      int k2u = mt * 16 + r + 8 * half;
      tg[(size_t)k2u * IMG_ + (pt * 16 + pl) * LEN_ + wt * 16 + wl] =
          bfc(acc[r]);
    }
  }
}

// --- stage 3: out[b,u,p,q] = sum_{k2,w} s[b,(k2,u),p,w] * cheb2b[k2][w][q] ---
// sS chunk (u=16 planes x p=16 rows x w=32 bf16 = 16 KB) is a strided 3D tile:
// one TENSOR_LOAD_TO_LDS per K-chunk, issued by wave 0, tracked by TENSORcnt.
__global__ void __launch_bounds__(256)
stage3_kernel(const __bf16* __restrict__ s, const __bf16* __restrict__ cheb2b,
              float* __restrict__ out) {
  __shared__ __bf16 sS[16 * 16 * 32];  // A chunk: [u][p][w_local]
  __shared__ __bf16 sB[128 * 32];      // B^T chunk: [q][w_local]
  int bid = blockIdx.x;
  int pt = bid & 7, b = bid >> 3;
  int tid = threadIdx.x, lane = tid & 31, wv = tid >> 5;
  int u0 = wv * 2;

  v8f acc[2][8];
#pragma unroll
  for (int du = 0; du < 2; ++du)
#pragma unroll
    for (int qt = 0; qt < 8; ++qt) acc[du][qt] = v8f_zero();

  const __bf16* sg = s + (size_t)b * KC_ * IMG_;
  for (int ch = 0; ch < 20; ++ch) {  // K = 5*128 = 640, chunks of 32 (w-major)
    int k2 = ch >> 2, w0 = (ch & 3) * 32;
    __syncthreads();
    if (wv == 0) {
      // Tensor DMA descriptor (ISA ch.8): 3D tile 32(w) x 16(p) x 16(u),
      // data_size=2B, strides 128 / 16384, dest = sS.
      unsigned long long ga = (unsigned long long)(size_t)(
          sg + (size_t)(k2 * 16) * IMG_ + (pt * 16) * LEN_ + w0);
      v4u g0;
      g0[0] = 1u;                                   // count=1
      g0[1] = lds_off(&sS[0]);                      // lds_addr
      g0[2] = (unsigned)(ga & 0xffffffffu);         // global_addr[31:0]
      g0[3] = (unsigned)((ga >> 32) & 0x01ffffffu)  // global_addr[56:32]
              | 0x80000000u;                        // type=2 (bits 127:126)
      v8u g1;
      g1[0] = 1u << 16;                     // data_size=1 -> 2 bytes
      g1[1] = 128u << 16;                   // tensor_dim0[15:0]=128
      g1[2] = 128u << 16;                   // dim0 hi=0 | tensor_dim1 lo=128
      g1[3] = 32u << 16;                    // dim1 hi=0 | tile_dim0=32
      g1[4] = 16u | (16u << 16);            // tile_dim1=16, tile_dim2=16
      g1[5] = 128u;                         // tensor_dim0_stride lo = 128
      g1[6] = 16384u << 16;                 // stride0 hi=0 | stride1 lo=16384
      g1[7] = 0u;                           // stride1 hi
      v4u g2;
      g2[0] = 16u;                          // tensor_dim2 = 16
      g2[1] = 0u;                           // tensor_dim3
      g2[2] = 16384u;                       // tensor_dim2_stride lo
      g2[3] = 0u;                           // stride hi | tile_dim3=0
      v4u g3 = {0u, 0u, 0u, 0u};
      asm volatile("tensor_load_to_lds %0, %1, %2, %3"
                   :: "s"(g0), "s"(g1), "s"(g2), "s"(g3) : "memory");
    }
    // B^T transpose copy stays in VALU, overlaps the TDM transfer.
    for (int i = tid; i < 128 * 32; i += 256) {
      int q = i >> 5, wl = i & 31;
      sB[i] = cheb2b[(size_t)k2 * IMG_ + (w0 + wl) * LEN_ + q];
    }
    if (wv == 0) __builtin_amdgcn_s_wait_tensorcnt(0);
    __syncthreads();
#pragma unroll
    for (int du = 0; du < 2; ++du) {
      v16bf af = load_a(&sS[(u0 + du) * 512], 32, lane);
#pragma unroll
      for (int qt = 0; qt < 8; ++qt) {
        v16bf bfr = load_bt(&sB[(qt * 16) * 32], 32, lane);
        acc[du][qt] = wmma_bf16(af, bfr, acc[du][qt]);
      }
    }
  }
  int n = lane & 15, half = lane >> 4;
#pragma unroll
  for (int du = 0; du < 2; ++du)
#pragma unroll
    for (int qt = 0; qt < 8; ++qt)
#pragma unroll
      for (int r = 0; r < 8; ++r)
        out[(((size_t)b * U_ + u0 + du) * LEN_ + pt * 16 + r + 8 * half) * LEN_ +
            qt * 16 + n] = acc[du][qt][r];
}

extern "C" void kernel_launch(void* const* d_in, const int* in_sizes, int n_in,
                              void* d_out, int out_size, void* d_ws,
                              size_t ws_size, hipStream_t stream) {
  const float* x     = (const float*)d_in[0];
  const float* coefs = (const float*)d_in[1];
  const float* cheb1 = (const float*)d_in[2];
  const float* cheb2 = (const float*)d_in[3];
  float* out = (float*)d_out;

  char* ws = (char*)d_ws;
  size_t off = 0;
  __bf16* t1 = (__bf16*)(ws + off);
  off += (size_t)B_ * KC_ * IMG_ * sizeof(__bf16);  // 83.9 MB (also holds s)
  __bf16* cheb1T = (__bf16*)(ws + off);
  off += (size_t)POW_ * IMG_ * sizeof(__bf16);
  __bf16* cheb2b = (__bf16*)(ws + off);
  off += (size_t)POW_ * IMG_ * sizeof(__bf16);
  __bf16* coefsA = (__bf16*)(ws + off);
  off += (size_t)KC_ * KC_ * sizeof(__bf16);

  prep_kernel<<<160, 256, 0, stream>>>(cheb1, cheb2, coefs, cheb1T, cheb2b,
                                       coefsA);
  stage1_kernel<<<B_ * POW_ * C_, 256, 0, stream>>>(x, cheb1T, t1);
  stage2_kernel<<<B_ * 64, 256, 0, stream>>>(t1, coefsA);
  stage3_kernel<<<B_ * 8, 256, 0, stream>>>(t1, cheb2b, out);
}
